// MoEMLP_23570780520542
// MI455X (gfx1250) — compile-verified, matrix operand order checked by reading, not run on hardware
//
#include <hip/hip_runtime.h>

// Problem constants (match reference): B=2, S=2048 -> T=4096 tokens
#define TOKENS 4096
#define DIM    1024   // D (n_embed)
#define MDIM   1408   // M (moe mlp)
#define NEXP   8      // E experts
// top-k = 2

typedef __attribute__((ext_vector_type(16))) __bf16          v16bf;
typedef __attribute__((ext_vector_type(16))) unsigned short  v16u16;
typedef __attribute__((ext_vector_type(8)))  float           v8f;

struct qpair { uint4 lo, hi; };   // 32B == one 16-element bf16 fragment

__device__ __forceinline__ unsigned short f32_to_bf16(float f) {
  unsigned int u = __float_as_uint(f);
  u += 0x7FFFu + ((u >> 16) & 1u);   // round-to-nearest-even
  return (unsigned short)(u >> 16);
}

__device__ __forceinline__ v16bf frag_from(const unsigned short* p) {
  qpair q;
  q.lo = *(const uint4*)(p);
  q.hi = *(const uint4*)(p + 16);
  return __builtin_bit_cast(v16bf, q);
}

// ---------------------------------------------------------------- utilities

__global__ void zero_counts_kernel(int* __restrict__ counts) {
  if (threadIdx.x < NEXP) counts[threadIdx.x] = 0;
}

__global__ void cvt_f32_to_bf16_kernel(const float* __restrict__ src,
                                       unsigned short* __restrict__ dst, int n) {
  int i = blockIdx.x * blockDim.x + threadIdx.x;
  int stride = gridDim.x * blockDim.x;
  for (; i < n; i += stride) dst[i] = f32_to_bf16(src[i]);
}

// Transpose-convert one expert slab: src [R][C] fp32 -> dst [C][R] bf16.
// grid: (C/32, R/32, E), block: 256.
__global__ __launch_bounds__(256)
void cvtT_f32_to_bf16_kernel(const float* __restrict__ src,
                             unsigned short* __restrict__ dst, int R, int C) {
  __shared__ unsigned short tile[32][33];
  const int e = blockIdx.z;
  const float* s = src + (size_t)e * R * C;
  unsigned short* d = dst + (size_t)e * R * C;
  const int c0 = blockIdx.x * 32, r0 = blockIdx.y * 32;
  const int tc = threadIdx.x & 31, tg = threadIdx.x >> 5;  // 8 groups x 4 rows
#pragma unroll
  for (int i = 0; i < 4; ++i) {
    int r = tg * 4 + i;
    tile[r][tc] = f32_to_bf16(s[(size_t)(r0 + r) * C + c0 + tc]);
  }
  __syncthreads();
#pragma unroll
  for (int i = 0; i < 4; ++i) {
    int r = tg * 4 + i;
    d[(size_t)(c0 + r) * R + r0 + tc] = tile[tc][r];
  }
}

// ---------------------------------------------------------------- router
// One wave32 per token: scores = x[t,:] @ w_gate (D x E), top-2, softmax over
// the 2 selected scores, build per-expert gather lists (slot id a = 2t+slot).
__global__ __launch_bounds__(256)
void router_kernel(const float* __restrict__ x, const float* __restrict__ wg,
                   float* __restrict__ prob, int* __restrict__ counts,
                   int* __restrict__ lists) {
  const int wid = threadIdx.x >> 5, lane = threadIdx.x & 31;
  const int t = blockIdx.x * 8 + wid;
  if (t >= TOKENS) return;

  float acc[NEXP];
#pragma unroll
  for (int e = 0; e < NEXP; ++e) acc[e] = 0.f;

  const float* xr = x + (size_t)t * DIM;
  for (int d = lane; d < DIM; d += 32) {
    float xv = xr[d];
    const float* wr = wg + (size_t)d * NEXP;
#pragma unroll
    for (int e = 0; e < NEXP; ++e) acc[e] = fmaf(xv, wr[e], acc[e]);
  }
#pragma unroll
  for (int off = 16; off; off >>= 1)
#pragma unroll
    for (int e = 0; e < NEXP; ++e) acc[e] += __shfl_xor(acc[e], off, 32);

  if (lane == 0) {
    int e0 = 0; float s0 = acc[0];
#pragma unroll
    for (int e = 1; e < NEXP; ++e) if (acc[e] > s0) { s0 = acc[e]; e0 = e; }
    int e1 = -1; float s1 = -3.0e38f;
#pragma unroll
    for (int e = 0; e < NEXP; ++e) if (e != e0 && acc[e] > s1) { s1 = acc[e]; e1 = e; }
    float p0 = 1.f / (1.f + __expf(s1 - s0));   // softmax over the two picks
    prob[2 * t]     = p0;
    prob[2 * t + 1] = 1.f - p0;
    int pos0 = atomicAdd(&counts[e0], 1); lists[e0 * TOKENS + pos0] = 2 * t;
    int pos1 = atomicAdd(&counts[e1], 1); lists[e1 * TOKENS + pos1] = 2 * t + 1;
  }
}

// ---------------------------------------------------------------- gate/up GEMM
// Tile: 16 gathered tokens x 128 M-cols; 8 waves, one 16x16 col-subtile each.
// A panel (16 x 1024 bf16, 32KB) staged in LDS once; B streamed per lane from
// N-major (transposed) weights -> every fragment is two 16B contiguous loads.
__global__ __launch_bounds__(256)
void moe_upgate_kernel(const unsigned short* __restrict__ xb,
                       const unsigned short* __restrict__ wgT,   // [E][M][D]
                       const unsigned short* __restrict__ wuT,   // [E][M][D]
                       const int* __restrict__ counts, const int* __restrict__ lists,
                       unsigned short* __restrict__ hidden) {
  const int e = blockIdx.z;
  const int cnt = counts[e];
  const int row0 = blockIdx.x * 16;
  if (row0 >= cnt) return;
  const int mbase = blockIdx.y * 128;

  __shared__ int s_a[16];
  __shared__ unsigned short s_x[16][DIM];   // 32 KB

  const int tid = threadIdx.x;
  if (tid < 16) {
    int r = row0 + tid;
    s_a[tid] = (r < cnt) ? lists[e * TOKENS + r] : -1;
  }
  __syncthreads();

  // stage gathered A panel: 16 rows x 1024 bf16 (8 x uint4 per thread)
  {
    const int srow = tid >> 4;
    const int qbase = (tid & 15) * 8;            // 128 uint4 per row
    uint4* drow = (uint4*)&s_x[srow][0];
    int a = s_a[srow];
    if (a >= 0) {
      const uint4* srow_g = (const uint4*)(xb + (size_t)(a >> 1) * DIM);
#pragma unroll
      for (int j = 0; j < 8; ++j) drow[qbase + j] = srow_g[qbase + j];
    } else {
      uint4 z = {0u, 0u, 0u, 0u};
#pragma unroll
      for (int j = 0; j < 8; ++j) drow[qbase + j] = z;
    }
  }
  __syncthreads();

  const int lane = tid & 31, wid = tid >> 5;
  const int frow  = lane & 15;            // A row / B,C col within 16
  const int khalf = (lane >> 4) << 3;     // K half select per ISA 7.12.2
  const int col = mbase + wid * 16 + frow;

  const unsigned short* bg = wgT + ((size_t)(e * MDIM + col)) * DIM;
  const unsigned short* bu = wuT + ((size_t)(e * MDIM + col)) * DIM;
  __builtin_prefetch(bg, 0, 0);
  __builtin_prefetch(bu, 0, 0);

  v8f cg = {};
  v8f cu = {};

  for (int kc = 0; kc < DIM; kc += 32) {
    v16bf a  = frag_from(&s_x[frow][kc + khalf]);
    v16bf fg = frag_from(bg + kc + khalf);
    v16bf fu = frag_from(bu + kc + khalf);
    cg = __builtin_amdgcn_wmma_f32_16x16x32_bf16(false, a, false, fg,
                                                 (short)0, cg, false, false);
    cu = __builtin_amdgcn_wmma_f32_16x16x32_bf16(false, a, false, fu,
                                                 (short)0, cu, false, false);
  }

  // SwiGLU, store bf16 hidden (C layout: VGPR i -> row i + 8*(lane>=16))
#pragma unroll
  for (int i = 0; i < 8; ++i) {
    int ri = i + ((lane >> 4) << 3);
    int ci = lane & 15;
    int a = s_a[ri];
    if (a >= 0) {
      float g = cg[i], u = cu[i];
      float h = (g / (1.f + __expf(-g))) * u;
      hidden[(size_t)a * MDIM + mbase + wid * 16 + ci] = f32_to_bf16(h);
    }
  }
}

// ---------------------------------------------------------------- down GEMM
// Tile: 16 gathered hidden rows x 256 D-cols; 8 waves x 2 16x16 subtiles.
// A panel (16 x 1408 bf16, 44KB) staged once; B streamed from Wd^T [E][D][M].
__global__ __launch_bounds__(256)
void moe_down_kernel(const unsigned short* __restrict__ hidden,
                     const unsigned short* __restrict__ wdT,     // [E][D][M]
                     const int* __restrict__ counts, const int* __restrict__ lists,
                     float* __restrict__ partial) {
  const int e = blockIdx.z;
  const int cnt = counts[e];
  const int row0 = blockIdx.x * 16;
  if (row0 >= cnt) return;
  const int dbase = blockIdx.y * 256;

  __shared__ int s_a[16];
  __shared__ unsigned short s_h[16][MDIM];  // 44 KB

  const int tid = threadIdx.x;
  if (tid < 16) {
    int r = row0 + tid;
    s_a[tid] = (r < cnt) ? lists[e * TOKENS + r] : -1;
  }
  __syncthreads();

  // stage gathered hidden panel: 16 rows x 1408 bf16 (11 x uint4 per thread)
  {
    const int srow = tid >> 4;
    const int qbase = (tid & 15) * 11;           // 176 uint4 per row
    uint4* drow = (uint4*)&s_h[srow][0];
    int a = s_a[srow];
    if (a >= 0) {
      const uint4* srow_g = (const uint4*)(hidden + (size_t)a * MDIM);
#pragma unroll
      for (int j = 0; j < 11; ++j) drow[qbase + j] = srow_g[qbase + j];
    } else {
      uint4 z = {0u, 0u, 0u, 0u};
#pragma unroll
      for (int j = 0; j < 11; ++j) drow[qbase + j] = z;
    }
  }
  __syncthreads();

  const int lane = tid & 31, wid = tid >> 5;
  const int frow  = lane & 15;
  const int khalf = (lane >> 4) << 3;
  const int col0 = dbase + wid * 32 + frow;

  const unsigned short* b0 = wdT + ((size_t)(e * DIM + col0)) * MDIM;
  const unsigned short* b1 = wdT + ((size_t)(e * DIM + col0 + 16)) * MDIM;
  __builtin_prefetch(b0, 0, 0);
  __builtin_prefetch(b1, 0, 0);

  v8f c0 = {};
  v8f c1 = {};

  for (int kc = 0; kc < MDIM; kc += 32) {
    v16bf a  = frag_from(&s_h[frow][kc + khalf]);
    v16bf f0 = frag_from(b0 + kc + khalf);
    v16bf f1 = frag_from(b1 + kc + khalf);
    c0 = __builtin_amdgcn_wmma_f32_16x16x32_bf16(false, a, false, f0,
                                                 (short)0, c0, false, false);
    c1 = __builtin_amdgcn_wmma_f32_16x16x32_bf16(false, a, false, f1,
                                                 (short)0, c1, false, false);
  }

#pragma unroll
  for (int i = 0; i < 8; ++i) {
    int ri = i + ((lane >> 4) << 3);
    int ci = lane & 15;
    int a = s_a[ri];
    if (a >= 0) {
      partial[(size_t)a * DIM + dbase + wid * 32 + ci]      = c0[i];
      partial[(size_t)a * DIM + dbase + wid * 32 + 16 + ci] = c1[i];
    }
  }
}

// ---------------------------------------------------------------- combine
// y[t] = p0 * partial[2t] + p1 * partial[2t+1]
__global__ __launch_bounds__(256)
void combine_kernel(const float* __restrict__ partial,
                    const float* __restrict__ prob, float* __restrict__ y) {
  const int QD = DIM / 4;
  int i = blockIdx.x * blockDim.x + threadIdx.x;
  if (i >= TOKENS * QD) return;
  int t = i / QD, c = i - t * QD;
  float p0 = prob[2 * t], p1 = prob[2 * t + 1];
  const float4* a0 = (const float4*)(partial + (size_t)(2 * t) * DIM);
  const float4* a1 = (const float4*)(partial + (size_t)(2 * t + 1) * DIM);
  float4 v0 = a0[c], v1 = a1[c];
  float4 r;
  r.x = p0 * v0.x + p1 * v1.x;
  r.y = p0 * v0.y + p1 * v1.y;
  r.z = p0 * v0.z + p1 * v1.z;
  r.w = p0 * v0.w + p1 * v1.w;
  ((float4*)y)[i] = r;
}

// ---------------------------------------------------------------- launch

extern "C" void kernel_launch(void* const* d_in, const int* in_sizes, int n_in,
                              void* d_out, int out_size, void* d_ws, size_t ws_size,
                              hipStream_t stream) {
  const float* x      = (const float*)d_in[0];   // [B,S,D]
  const float* w_gate = (const float*)d_in[1];   // [D,E]
  const float* w_g    = (const float*)d_in[2];   // [E,D,M]
  const float* w_u    = (const float*)d_in[3];   // [E,D,M]
  const float* w_d    = (const float*)d_in[4];   // [E,M,D]

  char* ws = (char*)d_ws;
  size_t off = 0;
  auto carve = [&](size_t bytes) -> char* {
    char* p = ws + off;
    off = (off + bytes + 255) & ~(size_t)255;
    return p;
  };
  int*   counts = (int*)carve(NEXP * sizeof(int));
  int*   lists  = (int*)carve((size_t)NEXP * TOKENS * sizeof(int));
  float* prob   = (float*)carve((size_t)2 * TOKENS * sizeof(float));
  unsigned short* xb  = (unsigned short*)carve((size_t)TOKENS * DIM * 2);
  unsigned short* wgT = (unsigned short*)carve((size_t)NEXP * DIM * MDIM * 2);
  unsigned short* wuT = (unsigned short*)carve((size_t)NEXP * DIM * MDIM * 2);
  unsigned short* wdT = (unsigned short*)carve((size_t)NEXP * MDIM * DIM * 2);
  unsigned short* hid = (unsigned short*)carve((size_t)2 * TOKENS * MDIM * 2);
  float* partial = (float*)carve((size_t)2 * TOKENS * DIM * sizeof(float));

  zero_counts_kernel<<<1, 32, 0, stream>>>(counts);

  cvt_f32_to_bf16_kernel<<<1024, 256, 0, stream>>>(x, xb, TOKENS * DIM);
  // w_g, w_u: [E][D][M] -> [E][M][D] bf16
  cvtT_f32_to_bf16_kernel<<<dim3(MDIM / 32, DIM / 32, NEXP), 256, 0, stream>>>(
      w_g, wgT, DIM, MDIM);
  cvtT_f32_to_bf16_kernel<<<dim3(MDIM / 32, DIM / 32, NEXP), 256, 0, stream>>>(
      w_u, wuT, DIM, MDIM);
  // w_d: [E][M][D] -> [E][D][M] bf16
  cvtT_f32_to_bf16_kernel<<<dim3(DIM / 32, MDIM / 32, NEXP), 256, 0, stream>>>(
      w_d, wdT, MDIM, DIM);

  router_kernel<<<TOKENS / 8, 256, 0, stream>>>(x, w_gate, prob, counts, lists);

  moe_upgate_kernel<<<dim3(TOKENS / 16, MDIM / 128, NEXP), 256, 0, stream>>>(
      xb, wgT, wuT, counts, lists, hid);

  moe_down_kernel<<<dim3(TOKENS / 16, DIM / 256, NEXP), 256, 0, stream>>>(
      hid, wdT, counts, lists, partial);

  combine_kernel<<<(TOKENS * (DIM / 4) + 255) / 256, 256, 0, stream>>>(
      partial, prob, (float*)d_out);
}
